// DeepseekV2LiteDecoderLayer_13675175870991
// MI455X (gfx1250) — matrix operand order repaired and checked by reading, not compiled, not run on hardware
//
#include <hip/hip_runtime.h>

typedef __bf16 bf16_t;
typedef __attribute__((ext_vector_type(16))) __bf16 v16bf;
typedef __attribute__((ext_vector_type(8)))  __bf16 v8bf;
typedef __attribute__((ext_vector_type(8)))  float  v8f;

static __device__ __forceinline__ v16bf cat16(v8bf lo, v8bf hi) {
  return __builtin_shufflevector(lo, hi, 0,1,2,3,4,5,6,7,8,9,10,11,12,13,14,15);
}
static __device__ __forceinline__ v8f wmma_bf16f32(v16bf a, v16bf b, v8f c) {
  // D = A(16x32 bf16) * B(32x16 bf16) + C(16x16 f32)
  return __builtin_amdgcn_wmma_f32_16x16x32_bf16(false, a, false, b, (short)0, c, false, false);
}

// gfx1250 async global->LDS copy (tracked by ASYNCcnt), 16 bytes per lane.
static __device__ __forceinline__ void async_ld_b128(void* lds, const void* gaddr) {
  asm volatile("global_load_async_to_lds_b128 %0, %1, off"
               :: "v"((unsigned int)(unsigned long long)lds), "v"(gaddr)
               : "memory");
}
static __device__ __forceinline__ void wait_async_le4() {
  asm volatile("s_wait_asynccnt 0x4" ::: "memory");
}
static __device__ __forceinline__ void wait_async_0() {
  asm volatile("s_wait_asynccnt 0x0" ::: "memory");
}

//======================= fp32 -> bf16 weight conversion (one pass per weight) ==========
__global__ __launch_bounds__(256)
void cvt_bf16_kernel(const float* __restrict__ in, bf16_t* __restrict__ out, long long n)
{
  long long i = ((long long)blockIdx.x * 256 + threadIdx.x) * 4;
  const long long stride = (long long)gridDim.x * 256 * 4;
  for (; i < n; i += stride) {
    __builtin_prefetch(in + i + 4096, 0, 1);   // global_prefetch_b8
    const float4 f = *(const float4*)(in + i);
    out[i + 0] = (bf16_t)f.x;
    out[i + 1] = (bf16_t)f.y;
    out[i + 2] = (bf16_t)f.z;
    out[i + 3] = (bf16_t)f.w;
  }
}

//======================= RMSNorm (fp32 or bf16 input -> bf16 out) =======================
__global__ __launch_bounds__(256)
void rmsnorm_kernel(const float* __restrict__ inF, const bf16_t* __restrict__ inB,
                    int stride, int cols, const float* __restrict__ w,
                    bf16_t* __restrict__ out)
{
  __shared__ float red[256];
  const int row = blockIdx.x;
  const int t = threadIdx.x;
  float ss = 0.f;
  for (int c = t; c < cols; c += 256) {
    float x = inF ? inF[(size_t)row * stride + c] : (float)inB[(size_t)row * stride + c];
    ss += x * x;
  }
  red[t] = ss;
  __syncthreads();
  for (int o = 128; o > 0; o >>= 1) {
    if (t < o) red[t] += red[t + o];
    __syncthreads();
  }
  const float r = rsqrtf(red[0] / (float)cols + 1e-6f);
  for (int c = t; c < cols; c += 256) {
    float x = inF ? inF[(size_t)row * stride + c] : (float)inB[(size_t)row * stride + c];
    out[(size_t)row * cols + c] = (bf16_t)(x * r * w[c]);
  }
}

//======================= WMMA GEMM: C[M,N] = A[M,K] @ W[N,K]^T (+resid) ================
// A, W: bf16 row-major. Double-buffered LDS staging via async global->LDS copies.
// 256 threads = 8 waves; block tile 128x128, K-step 32; wave tile 64x32 (4x2 WMMA tiles).
__global__ __launch_bounds__(256)
void gemm_bf16_kernel(const bf16_t* __restrict__ A, const bf16_t* __restrict__ W,
                      int M, int N, int K, const float* __restrict__ resid,
                      float* __restrict__ outF, bf16_t* __restrict__ outB)
{
  __shared__ __align__(16) bf16_t As[2][128 * 40];   // 128 rows x 32 k, padded to 40
  __shared__ __align__(16) bf16_t Bs[2][128 * 40];
  const int tid  = threadIdx.x;
  const int wid  = tid >> 5;
  const int lane = tid & 31;
  const int mbase = blockIdx.y * 128;
  const int nbase = blockIdx.x * 128;
  const int wm = (wid >> 2) * 64;   // wave M offset: 0 / 64
  const int wn = (wid & 3) * 32;    // wave N offset: 0/32/64/96
  const int fm   = lane & 15;
  const int half = lane >> 4;

  v8f zero = {};
  v8f acc[4][2];
  #pragma unroll
  for (int i = 0; i < 4; ++i)
    #pragma unroll
    for (int j = 0; j < 2; ++j) acc[i][j] = zero;

  const int lrow = tid >> 1;           // 0..127
  const int lcol = (tid & 1) * 16;     // 0 / 16
  int nrow = nbase + lrow;
  if (nrow >= N) nrow = N - 1;         // clamp: junk cols are masked at the store

  // Issue async copies of one 128x32 A tile + 128x32 W tile into buffer `buf`.
  auto issue = [&](int buf, int k0) {
    const bf16_t* ap = A + (size_t)(mbase + lrow) * K + k0 + lcol;
    bf16_t* la = &As[buf][lrow * 40 + lcol];
    async_ld_b128(la,     ap);
    async_ld_b128(la + 8, ap + 8);
    const bf16_t* wp = W + (size_t)nrow * K + k0 + lcol;
    bf16_t* lb = &Bs[buf][lrow * 40 + lcol];
    async_ld_b128(lb,     wp);
    async_ld_b128(lb + 8, wp + 8);
  };

  const int nsteps = K >> 5;
  issue(0, 0);
  for (int s = 0; s < nsteps; ++s) {
    const int cur = s & 1;
    const bool more = (s + 1) < nsteps;
    if (more) {
      issue(cur ^ 1, (s + 1) << 5);   // prefetch next tile while computing this one
      wait_async_le4();               // 4 newly issued outstanding -> current tile done
    } else {
      wait_async_0();
    }
    __syncthreads();

    // ---- fragments per ISA 7.12.2 layouts ----
    v16bf af[4];
    #pragma unroll
    for (int mi = 0; mi < 4; ++mi) {
      const int base = (wm + mi * 16 + fm) * 40;
      af[mi] = cat16(*(const v8bf*)&As[cur][base + half * 8],
                     *(const v8bf*)&As[cur][base + 16 + half * 8]);
    }
    v16bf bfm[2];
    #pragma unroll
    for (int ni = 0; ni < 2; ++ni) {
      const int base = (wn + ni * 16 + fm) * 40 + half * 8;
      bfm[ni] = cat16(*(const v8bf*)&Bs[cur][base], *(const v8bf*)&Bs[cur][base + 16]);
    }
    #pragma unroll
    for (int mi = 0; mi < 4; ++mi)
      #pragma unroll
      for (int ni = 0; ni < 2; ++ni)
        acc[mi][ni] = wmma_bf16f32(af[mi], bfm[ni], acc[mi][ni]);
    __syncthreads();   // all waves done reading `cur` before it is refilled
  }

  // ---- epilogue (C layout: row = v + 8*half, col = lane&15) ----
  #pragma unroll
  for (int mi = 0; mi < 4; ++mi) {
    #pragma unroll
    for (int ni = 0; ni < 2; ++ni) {
      #pragma unroll
      for (int v = 0; v < 8; ++v) {
        const int row = mbase + wm + mi * 16 + v + 8 * half;
        const int col = nbase + wn + ni * 16 + fm;
        if (col < N) {
          float val = acc[mi][ni][v];
          const size_t idx = (size_t)row * N + col;
          if (resid) val += resid[idx];
          if (outF) outF[idx] = val;
          else      outB[idx] = (bf16_t)val;
        }
      }
    }
  }
}

//======================= YaRN RoPE: q_pe in place, build kf =======================
__global__ __launch_bounds__(64)
void rope_build_kernel(const int* __restrict__ pos_ids, bf16_t* __restrict__ q,
                       const bf16_t* __restrict__ ckv, const bf16_t* __restrict__ kv,
                       bf16_t* __restrict__ kf)
{
  const int tok = blockIdx.x;
  const int j   = threadIdx.x;      // output rope dim 0..63
  const int j2  = j & 31;
  const float p = (float)pos_ids[tok];

  // yarn inverse frequency (mscale ratio == 1 since MSCALE == MSCALE_ALL)
  const float ar = (float)j2 * (1.0f / 32.0f);
  const float fe = powf(10000.0f, -ar);
  const float fi = fe * (1.0f / 40.0f);
  const float twoPi = 6.283185307179586f;
  const float denom = 2.0f * logf(10000.0f);
  float lo = floorf(64.0f * logf(4096.0f / (32.0f * twoPi)) / denom);
  float hi = ceilf (64.0f * logf(4096.0f / ( 1.0f * twoPi)) / denom);
  hi = fminf(hi, 63.0f);
  if (hi == lo) hi += 0.001f;
  float ramp = ((float)j2 - lo) / (hi - lo);
  ramp = fminf(fmaxf(ramp, 0.0f), 1.0f);
  const float maskf = 1.0f - ramp;
  const float invf  = fi * (1.0f - maskf) + fe * maskf;
  const float ang = p * invf;
  const float cs = cosf(ang);
  const float sn = sinf(ang);

  // k_pe (deinterleave + rotate-half)
  const bf16_t* kp = ckv + (size_t)tok * 576 + 512;
  const float kx0 = (float)kp[2 * j2];
  const float kx1 = (float)kp[2 * j2 + 1];
  const float kout = (j < 32) ? (kx0 * cs - kx1 * sn) : (kx1 * cs + kx0 * sn);

  // q_pe, 16 heads, in place (read all, sync, write all)
  float qout[16];
  #pragma unroll
  for (int h = 0; h < 16; ++h) {
    const bf16_t* qp = q + (size_t)tok * 3072 + h * 192 + 128;
    const float a = (float)qp[2 * j2];
    const float b = (float)qp[2 * j2 + 1];
    qout[h] = (j < 32) ? (a * cs - b * sn) : (b * cs + a * sn);
  }
  __syncthreads();
  #pragma unroll
  for (int h = 0; h < 16; ++h)
    q[(size_t)tok * 3072 + h * 192 + 128 + j] = (bf16_t)qout[h];
  #pragma unroll
  for (int h = 0; h < 16; ++h)
    kf[(size_t)tok * 3072 + h * 192 + 128 + j] = (bf16_t)kout;

  // copy k_nope into kf (2048 bf16 per token, 32 per thread)
  for (int t = 0; t < 32; ++t) {
    const int idx = j * 32 + t;
    const int h = idx >> 7;
    const int d = idx & 127;
    kf[(size_t)tok * 3072 + h * 192 + d] = kv[(size_t)tok * 4096 + h * 256 + d];
  }
}

//======================= Flash attention (causal), WMMA QK^T and PV =======================
// grid: (B*16 heads, S/64).  128 threads = 4 independent waves, 16 query rows each.
__global__ __launch_bounds__(128)
void attn_kernel(const bf16_t* __restrict__ qf, const bf16_t* __restrict__ kf,
                 const bf16_t* __restrict__ kv, bf16_t* __restrict__ attn_out)
{
  __shared__ __align__(16) bf16_t Vt[4][128 * 40];  // per-wave V^T staging (dv x 32 keys)
  __shared__ __align__(16) bf16_t Pt[4][16 * 40];   // per-wave P (16 q x 32 keys)
  const int b   = blockIdx.x >> 4;
  const int h   = blockIdx.x & 15;
  const int wid = threadIdx.x >> 5;
  const int lane = threadIdx.x & 31;
  const int q0  = blockIdx.y * 64 + wid * 16;
  const int fm   = lane & 15;
  const int half = lane >> 4;
  bf16_t* vt = Vt[wid];
  bf16_t* pt = Pt[wid];

  // Q fragments (A 16x32 layout), rows q0+fm, d = 0..191 in 6 chunks
  v16bf qa[6];
  {
    const bf16_t* qp = qf + ((size_t)(b * 2048 + q0 + fm) * 16 + h) * 192;
    #pragma unroll
    for (int c = 0; c < 6; ++c) {
      const int d0 = c * 32 + half * 8;
      qa[c] = cat16(*(const v8bf*)(qp + d0), *(const v8bf*)(qp + d0 + 16));
    }
  }

  v8f zero = {};
  v8f oacc[8];
  #pragma unroll
  for (int t = 0; t < 8; ++t) oacc[t] = zero;
  float mrow[8], lrow[8];
  #pragma unroll
  for (int v = 0; v < 8; ++v) { mrow[v] = -1e30f; lrow[v] = 0.f; }

  const int nchunks = ((q0 + 15) >> 5) + 1;
  const float scale = 0.07216878364870322f;  // 192^-0.5

  for (int ck = 0; ck < nchunks; ++ck) {
    const int kc = ck * 32;

    // stage V^T: lane handles key kc+lane, dv 0..127
    {
      const bf16_t* vp = kv + ((size_t)(b * 2048 + kc + lane) * 16 + h) * 256 + 128;
      #pragma unroll
      for (int d = 0; d < 128; d += 8) {
        v8bf vv = *(const v8bf*)(vp + d);
        #pragma unroll
        for (int e = 0; e < 8; ++e) vt[(d + e) * 40 + lane] = vv[e];
      }
    }

    // scores: S = Q K^T over d=192 (two N-tiles of 16 keys)
    v8f s0 = zero, s1 = zero;
    #pragma unroll
    for (int c = 0; c < 6; ++c) {
      const int doff = c * 32 + half * 8;
      const bf16_t* kp0 = kf + ((size_t)(b * 2048 + kc + fm) * 16 + h) * 192 + doff;
      const bf16_t* kp1 = kf + ((size_t)(b * 2048 + kc + 16 + fm) * 16 + h) * 192 + doff;
      v16bf kb0 = cat16(*(const v8bf*)kp0, *(const v8bf*)(kp0 + 16));
      v16bf kb1 = cat16(*(const v8bf*)kp1, *(const v8bf*)(kp1 + 16));
      s0 = wmma_bf16f32(qa[c], kb0, s0);
      s1 = wmma_bf16f32(qa[c], kb1, s1);
    }

    // online softmax (row = v + 8*half, cols across 16 lanes of this half)
    const int col0 = kc + fm;
    const int col1 = col0 + 16;
    #pragma unroll
    for (int v = 0; v < 8; ++v) {
      const int qg = q0 + v + 8 * half;
      float a0 = (col0 <= qg) ? s0[v] * scale : -1e30f;
      float a1 = (col1 <= qg) ? s1[v] * scale : -1e30f;
      float mx = fmaxf(a0, a1);
      #pragma unroll
      for (int msk = 1; msk < 16; msk <<= 1)
        mx = fmaxf(mx, __shfl_xor(mx, msk, 32));
      const float mnew  = fmaxf(mrow[v], mx);
      const float alpha = __expf(mrow[v] - mnew);
      const float p0 = __expf(a0 - mnew);
      const float p1 = __expf(a1 - mnew);
      float rs = p0 + p1;
      #pragma unroll
      for (int msk = 1; msk < 16; msk <<= 1)
        rs += __shfl_xor(rs, msk, 32);
      lrow[v] = lrow[v] * alpha + rs;
      mrow[v] = mnew;
      const int prow = v + 8 * half;
      pt[prow * 40 + fm]      = (bf16_t)p0;
      pt[prow * 40 + fm + 16] = (bf16_t)p1;
      #pragma unroll
      for (int t = 0; t < 8; ++t) oacc[t][v] *= alpha;
    }

    // O += P V : A = P(16x32), B = V(32 keys x 16 dv) from transposed staging
    v16bf pa = cat16(*(const v8bf*)&pt[fm * 40 + half * 8],
                     *(const v8bf*)&pt[fm * 40 + 16 + half * 8]);
    #pragma unroll
    for (int t = 0; t < 8; ++t) {
      const int base = (t * 16 + fm) * 40 + half * 8;
      v16bf vb = cat16(*(const v8bf*)&vt[base], *(const v8bf*)&vt[base + 16]);
      oacc[t] = wmma_bf16f32(pa, vb, oacc[t]);
    }
  }

  // normalize and write bf16 output [b, s, h, dv]
  #pragma unroll
  for (int v = 0; v < 8; ++v) {
    const float inv = 1.0f / lrow[v];
    const int qg = q0 + v + 8 * half;
    bf16_t* op = attn_out + ((size_t)(b * 2048 + qg) * 16 + h) * 128;
    #pragma unroll
    for (int t = 0; t < 8; ++t)
      op[t * 16 + fm] = (bf16_t)(oacc[t][v] * inv);
  }
}

//======================= SiLU(g) * u =======================
__global__ __launch_bounds__(256)
void silu_mul_kernel(const bf16_t* __restrict__ g, const bf16_t* __restrict__ u,
                     bf16_t* __restrict__ out, long long n)
{
  long long i = (long long)blockIdx.x * 256 + threadIdx.x;
  const long long stride = (long long)gridDim.x * 256;
  for (; i < n; i += stride) {
    const float gv = (float)g[i];
    const float uv = (float)u[i];
    out[i] = (bf16_t)(gv / (1.0f + __expf(-gv)) * uv);
  }
}

//======================= host: pipeline =======================
extern "C" void kernel_launch(void* const* d_in, const int* in_sizes, int n_in,
                              void* d_out, int out_size, void* d_ws, size_t ws_size,
                              hipStream_t stream) {
  (void)in_sizes; (void)n_in; (void)out_size; (void)ws_size;
  const float* hidden = (const float*)d_in[0];
  const int*   pos    = (const int*)d_in[1];
  const float* Wq     = (const float*)d_in[2];
  const float* Wkva   = (const float*)d_in[3];
  const float* w_kvln = (const float*)d_in[4];
  const float* Wkvb   = (const float*)d_in[5];
  const float* Wo     = (const float*)d_in[6];
  const float* Wg     = (const float*)d_in[7];
  const float* Wu     = (const float*)d_in[8];
  const float* Wd     = (const float*)d_in[9];
  const float* w_ln1  = (const float*)d_in[10];
  const float* w_ln2  = (const float*)d_in[11];
  float* out = (float*)d_out;   // also the fp32 post-attention residual buffer

  char* ws = (char*)d_ws;
  size_t off = 0;
  auto wsalloc = [&](size_t bytes) -> char* {
    char* p = ws + off;
    off += (bytes + 255) & ~(size_t)255;
    return p;
  };
  // activations (bf16)
  bf16_t* xln  = (bf16_t*)wsalloc((size_t)4096 * 2048  * 2);
  bf16_t* q    = (bf16_t*)wsalloc((size_t)4096 * 3072  * 2);
  bf16_t* ckv  = (bf16_t*)wsalloc((size_t)4096 * 576   * 2);
  bf16_t* cln  = (bf16_t*)wsalloc((size_t)4096 * 512   * 2);
  bf16_t* kvb  = (bf16_t*)wsalloc((size_t)4096 * 4096  * 2);
  bf16_t* kfb  = (bf16_t*)wsalloc((size_t)4096 * 3072  * 2);
  bf16_t* attn = (bf16_t*)wsalloc((size_t)4096 * 2048  * 2);
  bf16_t* y    = (bf16_t*)wsalloc((size_t)4096 * 2048  * 2);
  bf16_t* g    = (bf16_t*)wsalloc((size_t)4096 * 10944 * 2);
  bf16_t* u    = (bf16_t*)wsalloc((size_t)4096 * 10944 * 2);
  // bf16 weights (~162 MB -> whole weight set fits in the 192 MB L2)
  bf16_t* Wq_b   = (bf16_t*)wsalloc((size_t)3072 * 2048  * 2);
  bf16_t* Wkva_b = (bf16_t*)wsalloc((size_t)576  * 2048  * 2);
  bf16_t* Wkvb_b = (bf16_t*)wsalloc((size_t)4096 * 512   * 2);
  bf16_t* Wo_b   = (bf16_t*)wsalloc((size_t)2048 * 2048  * 2);
  bf16_t* Wg_b   = (bf16_t*)wsalloc((size_t)10944 * 2048 * 2);
  bf16_t* Wu_b   = (bf16_t*)wsalloc((size_t)10944 * 2048 * 2);
  bf16_t* Wd_b   = (bf16_t*)wsalloc((size_t)2048 * 10944 * 2);

  // 0. weight down-conversion (once per call; deterministic)
  cvt_bf16_kernel<<<2048, 256, 0, stream>>>(Wq,   Wq_b,   (long long)3072 * 2048);
  cvt_bf16_kernel<<<1024, 256, 0, stream>>>(Wkva, Wkva_b, (long long)576  * 2048);
  cvt_bf16_kernel<<<1024, 256, 0, stream>>>(Wkvb, Wkvb_b, (long long)4096 * 512);
  cvt_bf16_kernel<<<2048, 256, 0, stream>>>(Wo,   Wo_b,   (long long)2048 * 2048);
  cvt_bf16_kernel<<<4096, 256, 0, stream>>>(Wg,   Wg_b,   (long long)10944 * 2048);
  cvt_bf16_kernel<<<4096, 256, 0, stream>>>(Wu,   Wu_b,   (long long)10944 * 2048);
  cvt_bf16_kernel<<<4096, 256, 0, stream>>>(Wd,   Wd_b,   (long long)2048 * 10944);

  // 1. ln1
  rmsnorm_kernel<<<4096, 256, 0, stream>>>(hidden, nullptr, 2048, 2048, w_ln1, xln);
  // 2. q = xln @ Wq^T            [4096,3072]
  gemm_bf16_kernel<<<dim3(24, 32), 256, 0, stream>>>(xln, Wq_b, 4096, 3072, 2048, nullptr, nullptr, q);
  // 3. ckv = xln @ Wkva^T        [4096,576]
  gemm_bf16_kernel<<<dim3(5, 32), 256, 0, stream>>>(xln, Wkva_b, 4096, 576, 2048, nullptr, nullptr, ckv);
  // 4. c_ln = rms(ckv[:,:512])
  rmsnorm_kernel<<<4096, 256, 0, stream>>>(nullptr, ckv, 576, 512, w_kvln, cln);
  // 5. kv = c_ln @ Wkvb^T        [4096,4096]
  gemm_bf16_kernel<<<dim3(32, 32), 256, 0, stream>>>(cln, Wkvb_b, 4096, 4096, 512, nullptr, nullptr, kvb);
  // 6. rope q_pe in place; build kf
  rope_build_kernel<<<4096, 64, 0, stream>>>(pos, q, ckv, kvb, kfb);
  // 7. attention -> attn [4096, 16*128]
  attn_kernel<<<dim3(32, 32), 128, 0, stream>>>(q, kfb, kvb, attn);
  // 8. out = hidden + attn @ Wo^T   (fp32 residual kept in d_out)
  gemm_bf16_kernel<<<dim3(16, 32), 256, 0, stream>>>(attn, Wo_b, 4096, 2048, 2048, hidden, out, nullptr);
  // 9. ln2
  rmsnorm_kernel<<<4096, 256, 0, stream>>>(out, nullptr, 2048, 2048, w_ln2, y);
  // 10. g = y @ Wg^T, u = y @ Wu^T   [4096,10944]
  gemm_bf16_kernel<<<dim3(86, 32), 256, 0, stream>>>(y, Wg_b, 4096, 10944, 2048, nullptr, nullptr, g);
  gemm_bf16_kernel<<<dim3(86, 32), 256, 0, stream>>>(y, Wu_b, 4096, 10944, 2048, nullptr, nullptr, u);
  // 11. h = silu(g)*u  (overwrites g)
  silu_mul_kernel<<<4096, 256, 0, stream>>>(g, u, g, (long long)4096 * 10944);
  // 12. out = out + h @ Wd^T
  gemm_bf16_kernel<<<dim3(16, 32), 256, 0, stream>>>(g, Wd_b, 4096, 2048, 10944, out, out, nullptr);
}
